// BrainNATLayer_60593398612311
// MI455X (gfx1250) — matrix operand order repaired and checked by reading, not compiled, not run on hardware
//
#include <hip/hip_runtime.h>
#include <hip/hip_bf16.h>
#include <stdint.h>

// ---------------- problem dims ----------------
#define B_   4
#define N_   2048
#define C_   256
#define H_   8
#define DH_  32
#define KTOP 64
#define R_   256
#define HALF 1024              // N/2
#define UNM  768               // HALF - R
#define NM   1792              // N - R  (tokens after merge)
#define HID_ 1024
#define M1   (B_ * N_)         // 8192 rows, phase-1 GEMMs
#define M2   (B_ * NM)         // 7168 rows, phase-2 GEMMs

// GEMM tiling
#define TM   32                // block tile M (2 waves)
#define TN   256               // block tile N (4 waves x 64)
#define TK   32                // k step per WMMA pass
#define LSTR 40                // ushort elems per 32-K row slice in LDS (64B data + 16B pad)

typedef __attribute__((ext_vector_type(16))) __bf16 v16bf;
typedef __attribute__((ext_vector_type(8)))  __bf16 v8bf;
typedef __attribute__((ext_vector_type(8)))  float  v8f;
typedef __attribute__((ext_vector_type(4)))  int    v4i;

#if __has_builtin(__builtin_amdgcn_global_load_async_to_lds_b128)
#define HAVE_ASYNC_LDS 1
#else
#define HAVE_ASYNC_LDS 0
#endif

// ---------------- helpers ----------------
__device__ __forceinline__ unsigned short f32_to_bf16_rn(float f) {
    unsigned int u = __float_as_uint(f);
    unsigned int lsb = (u >> 16) & 1u;
    u += 0x7fffu + lsb;
    return (unsigned short)(u >> 16);
}

__device__ __forceinline__ float gelu_tanh(float x) {
    float x3 = x * x * x;
    return 0.5f * x * (1.0f + tanhf(0.7978845608028654f * (x + 0.044715f * x3)));
}

__device__ __forceinline__ void sched_fence() {
#if __has_builtin(__builtin_amdgcn_sched_barrier)
    __builtin_amdgcn_sched_barrier(0);   // nothing may be reordered across this point
#endif
}

// 16-byte global -> LDS copy (async DMA when available)
__device__ __forceinline__ void cp16_g2l(const unsigned short* g, unsigned short* l) {
#if HAVE_ASYNC_LDS
    __builtin_amdgcn_global_load_async_to_lds_b128(
        (__attribute__((address_space(1))) v4i*)g,
        (__attribute__((address_space(3))) v4i*)l, 0, 0);
#else
    *reinterpret_cast<v8bf*>(l) = *reinterpret_cast<const v8bf*>(g);
#endif
}

__device__ __forceinline__ void wait_async_all() {
#if HAVE_ASYNC_LDS
#if __has_builtin(__builtin_amdgcn_s_wait_asynccnt)
    __builtin_amdgcn_s_wait_asynccnt(0);
#else
    asm volatile("s_wait_asynccnt 0" ::: "memory");
#endif
#endif
}

// ---------------- weight convert + transpose: (K x N) f32 -> (N x K) bf16 ----------------
__global__ void k_transpose_bf16(const float* __restrict__ in, unsigned short* __restrict__ out,
                                 int K, int Nn) {
    int total = K * Nn;
    for (int idx = blockIdx.x * blockDim.x + threadIdx.x; idx < total;
         idx += gridDim.x * blockDim.x) {
        int k = idx / Nn, n = idx % Nn;
        out[(size_t)n * K + k] = f32_to_bf16_rn(in[idx]);
    }
}

// ---------------- layernorm over C=256, one block per token, out bf16 ----------------
__global__ void k_layernorm_bf16(const float* __restrict__ x, const float* __restrict__ g,
                                 const float* __restrict__ bta, unsigned short* __restrict__ out) {
    __shared__ float red[C_];
    int tok = blockIdx.x, c = threadIdx.x;
    float v = x[(size_t)tok * C_ + c];
    red[c] = v; __syncthreads();
    #pragma unroll
    for (int off = C_ / 2; off > 0; off >>= 1) {
        if (c < off) red[c] += red[c + off];
        __syncthreads();
    }
    float mean = red[0] * (1.0f / C_);
    __syncthreads();
    float d = v - mean;
    red[c] = d * d; __syncthreads();
    #pragma unroll
    for (int off = C_ / 2; off > 0; off >>= 1) {
        if (c < off) red[c] += red[c + off];
        __syncthreads();
    }
    float var = red[0] * (1.0f / C_);
    float y = d * rsqrtf(var + 1e-5f) * g[c] + bta[c];
    out[(size_t)tok * C_ + c] = f32_to_bf16_rn(y);
}

// ---------------- WMMA bf16 GEMM with async-LDS double buffering ----------------
// out(MxN) = A(MxK,bf16) @ BT(NxK,bf16)^T + bias  [+res] [gelu] -> f32 or bf16
// Block: 256 thr = 8 waves (2 M-waves x 4 N-waves); block tile 32x256; wave tile 16x64.
// LDS tiles padded to 80B row stride so 16 lanes hit distinct banks (stride 20 words mod 64).
// Pipeline per K-step: wait(asynccnt buf_i) -> barrier -> async-issue buf_{i+1} ->
//   all fragment ds_loads -> sched_barrier -> one dscnt wait + 4 back-to-back WMMAs.
template <int GELU, int OUT_BF16, int HAS_RES>
__global__ __launch_bounds__(256)
void k_gemm_nt_wmma(const unsigned short* __restrict__ A,
                    const unsigned short* __restrict__ BT,
                    const float* __restrict__ bias,
                    const float* __restrict__ res,
                    float* __restrict__ outF,
                    unsigned short* __restrict__ outH,
                    int M, int Nn, int K) {
    __shared__ __align__(16) unsigned short Ab[2][TM * LSTR];   //  5.1 KB
    __shared__ __align__(16) unsigned short Bb[2][TN * LSTR];   // 41.0 KB

    const int tid  = threadIdx.x;
    const int lane = tid & 31;
    const int wave = tid >> 5;
    const int mw   = wave >> 2;          // 0..1
    const int nw   = wave & 3;           // 0..3
    const int m0   = blockIdx.y * TM;
    const int n0   = blockIdx.x * TN;
    if (n0 >= Nn) return;

    const int row = lane & 15;
    const int hi  = lane >> 4;

    // staging assignment: A tile = 128 16B-chunks (tid<128), B tile = 1024 chunks (4/thread)
    const int ar = tid >> 2;             // A row   (tid<128 -> 0..31)
    const int ac = tid & 3;              // A 16B part within 64B row slice

    v8f zero = {0.f, 0.f, 0.f, 0.f, 0.f, 0.f, 0.f, 0.f};
    v8f acc[4] = {zero, zero, zero, zero};

    auto stage = [&](int buf, int k0) {
        if (tid < 128) {
            cp16_g2l(A + (size_t)(m0 + ar) * K + k0 + ac * 8,
                     &Ab[buf][ar * LSTR + ac * 8]);
        }
        #pragma unroll
        for (int j = 0; j < 4; ++j) {
            int chunk = tid + j * 256;
            int col = chunk >> 2, cp = chunk & 3;
            cp16_g2l(BT + (size_t)(n0 + col) * K + k0 + cp * 8,
                     &Bb[buf][col * LSTR + cp * 8]);
        }
    };

    stage(0, 0);
    const int T = K / TK;
    for (int i = 0; i < T; ++i) {
        wait_async_all();
        __syncthreads();
        if (i + 1 < T) stage((i + 1) & 1, (i + 1) * TK);

        const int buf = i & 1;
        // ---- all fragment loads (1x A + 4x B) issued before any WMMA ----
        const unsigned short* ab = &Ab[buf][(mw * 16 + row) * LSTR];
        v8bf a0 = *reinterpret_cast<const v8bf*>(ab + hi * 8);
        v8bf a1 = *reinterpret_cast<const v8bf*>(ab + 16 + hi * 8);
        v8bf y0[4], y1[4];
        #pragma unroll
        for (int t = 0; t < 4; ++t) {
            const unsigned short* bb = &Bb[buf][(nw * 64 + t * 16 + row) * LSTR + hi * 8];
            y0[t] = *reinterpret_cast<const v8bf*>(bb);
            y1[t] = *reinterpret_cast<const v8bf*>(bb + 16);
        }
        sched_fence();   // keep ds_loads above, WMMAs below: 1 dscnt wait, 4 WMMAs back-to-back

        v16bf afrag;
        #pragma unroll
        for (int e = 0; e < 8; ++e) { afrag[e] = a0[e]; afrag[8 + e] = a1[e]; }

        #pragma unroll
        for (int t = 0; t < 4; ++t) {
            v16bf bfrag;
            #pragma unroll
            for (int e = 0; e < 8; ++e) { bfrag[e] = y0[t][e]; bfrag[8 + e] = y1[t][e]; }
            acc[t] = __builtin_amdgcn_wmma_f32_16x16x32_bf16(
                false, afrag, false, bfrag, (short)0, acc[t], false, false);
        }
    }

    // ---- epilogue: D layout lane=16*hi+n, VGPR v -> M = v + 8*hi ----
    #pragma unroll
    for (int t = 0; t < 4; ++t) {
        int ncol = n0 + nw * 64 + t * 16 + row;
        float bcol = bias[ncol];
        #pragma unroll
        for (int v = 0; v < 8; ++v) {
            int m = m0 + mw * 16 + v + 8 * hi;
            float val = acc[t][v] + bcol;
            if (HAS_RES) val += res[(size_t)m * Nn + ncol];
            if (GELU)    val = gelu_tanh(val);
            if (OUT_BF16) outH[(size_t)m * Nn + ncol] = f32_to_bf16_rn(val);
            else          outF[(size_t)m * Nn + ncol] = val;
        }
    }
}

// ---------------- top-k attention: one block (128 thr) per (b,h,n) ----------------
__global__ void k_topk_attn(const float* __restrict__ qkv, const unsigned char* __restrict__ mask,
                            unsigned short* __restrict__ o_bf16) {
    __shared__ float q[DH_];
    __shared__ float sc[N_];
    __shared__ float rv[128];
    __shared__ int   ri[128];
    __shared__ float topv[KTOP];
    __shared__ int   topi[KTOP];
    __shared__ float p[KTOP];
    __shared__ float ssum;

    int bid = blockIdx.x;
    int n = bid % N_;
    int h = (bid / N_) % H_;
    int b = bid / (N_ * H_);
    int t = threadIdx.x;

    size_t qbase = ((size_t)b * N_ + n) * (3 * C_) + 0 * C_ + h * DH_;
    if (t < DH_) q[t] = qkv[qbase + t];
    __syncthreads();

    for (int m = t; m < N_; m += 128) {
        float s;
        if (mask[(size_t)b * N_ + m]) {
            size_t kb = ((size_t)b * N_ + m) * (3 * C_) + 1 * C_ + h * DH_;
            float acc = 0.f;
            #pragma unroll
            for (int d = 0; d < DH_; ++d) acc += q[d] * qkv[kb + d];
            s = acc * 0.17677669529663687f;  // Dh^-0.5
        } else {
            s = -3.402823466e38f;            // finfo(f32).min
        }
        sc[m] = s;
    }
    __syncthreads();

    // iterative top-64 extraction (ties -> lowest index, matches lax.top_k)
    for (int ksel = 0; ksel < KTOP; ++ksel) {
        float bv = -__builtin_inff();
        int   bi = 0x7fffffff;
        for (int m = t; m < N_; m += 128) {
            float v = sc[m];
            if (v > bv || (v == bv && m < bi)) { bv = v; bi = m; }
        }
        rv[t] = bv; ri[t] = bi; __syncthreads();
        #pragma unroll
        for (int off = 64; off > 0; off >>= 1) {
            if (t < off) {
                float v2 = rv[t + off]; int i2 = ri[t + off];
                if (v2 > rv[t] || (v2 == rv[t] && i2 < ri[t])) { rv[t] = v2; ri[t] = i2; }
            }
            __syncthreads();
        }
        if (t == 0) { topv[ksel] = rv[0]; topi[ksel] = ri[0]; sc[ri[0]] = -__builtin_inff(); }
        __syncthreads();
    }

    if (t < KTOP) p[t] = expf(topv[t] - topv[0]);
    __syncthreads();
    if (t == 0) {
        float s = 0.f;
        for (int i = 0; i < KTOP; ++i) s += p[i];
        ssum = s;
    }
    __syncthreads();

    if (t < DH_) {
        float acc = 0.f;
        for (int k2 = 0; k2 < KTOP; ++k2) {
            size_t vb = ((size_t)b * N_ + topi[k2]) * (3 * C_) + 2 * C_ + h * DH_;
            acc += p[k2] * qkv[vb + t];
        }
        acc /= ssum;
        o_bf16[((size_t)b * N_ + n) * C_ + h * DH_ + t] = f32_to_bf16_rn(acc);
    }
}

// ---------------- metric = mean_h(k), L2-normalized. one block (32 thr) per (b,n) ----------------
__global__ void k_metric(const float* __restrict__ qkv, float* __restrict__ mnorm) {
    __shared__ float mm[DH_];
    int bn = blockIdx.x;
    int d = threadIdx.x;
    size_t base = (size_t)bn * (3 * C_) + 1 * C_ + d;
    float s = 0.f;
    #pragma unroll
    for (int h = 0; h < H_; ++h) s += qkv[base + h * DH_];
    s *= (1.0f / H_);
    mm[d] = s; __syncthreads();
    float nrm = 0.f;
    #pragma unroll
    for (int i = 0; i < DH_; ++i) nrm += mm[i] * mm[i];
    nrm = sqrtf(nrm) + 1e-6f;
    mnorm[(size_t)bn * DH_ + d] = s / nrm;
}

// ---------------- sim row argmax: one block (128 thr) per (b,i) even-token row ----------------
__global__ void k_simmax(const float* __restrict__ mnorm, float* __restrict__ node_max,
                         int* __restrict__ node_idx) {
    __shared__ float av[DH_];
    __shared__ float rv[128];
    __shared__ int   ri[128];
    int bi = blockIdx.x;
    int i = bi % HALF, b = bi / HALF;
    int t = threadIdx.x;
    const float* Arow = mnorm + ((size_t)b * N_ + 2 * i) * DH_;
    if (t < DH_) av[t] = Arow[t];
    __syncthreads();
    float bv = -__builtin_inff();
    int bidx = 0x7fffffff;
    for (int j = t; j < HALF; j += 128) {
        const float* Brow = mnorm + ((size_t)b * N_ + 2 * j + 1) * DH_;
        float s = 0.f;
        #pragma unroll
        for (int d = 0; d < DH_; ++d) s += av[d] * Brow[d];
        if (s > bv || (s == bv && j < bidx)) { bv = s; bidx = j; }
    }
    rv[t] = bv; ri[t] = bidx; __syncthreads();
    #pragma unroll
    for (int off = 64; off > 0; off >>= 1) {
        if (t < off) {
            float v2 = rv[t + off]; int i2 = ri[t + off];
            if (v2 > rv[t] || (v2 == rv[t] && i2 < ri[t])) { rv[t] = v2; ri[t] = i2; }
        }
        __syncthreads();
    }
    if (t == 0) { node_max[bi] = rv[0]; node_idx[bi] = ri[0]; }
}

// ---------------- stable descending argsort via rank counting: one block (1024 thr) per batch ----
__global__ void k_rank(const float* __restrict__ node_max, int* __restrict__ edge_idx) {
    __shared__ float vals[HALF];
    int b = blockIdx.x, i = threadIdx.x;
    float v = node_max[(size_t)b * HALF + i];
    vals[i] = v; __syncthreads();
    int r = 0;
    for (int j = 0; j < HALF; ++j) {
        float vj = vals[j];
        if (vj > v || (vj == v && j < i)) ++r;
    }
    edge_idx[(size_t)b * HALF + r] = i;
}

// ---------------- merge: unmerged rows (first 768 output tokens) ----------------
__global__ void k_merge_unm(const float* __restrict__ xres, const int* __restrict__ edge,
                            float* __restrict__ xm) {
    int bt = blockIdx.x;
    int t = bt % UNM, b = bt / UNM, c = threadIdx.x;
    int idx = edge[(size_t)b * HALF + R_ + t];  // unm_idx
    xm[((size_t)b * NM + t) * C_ + c] = xres[((size_t)b * N_ + 2 * idx) * C_ + c];
}

// ---------------- merge: dst rows with deterministic in-order accumulation ----------------
__global__ void k_merge_dst(const float* __restrict__ xres, const int* __restrict__ edge,
                            const int* __restrict__ node_idx, float* __restrict__ xm) {
    int bj = blockIdx.x;
    int j = bj % HALF, b = bj / HALF, c = threadIdx.x;
    float val = xres[((size_t)b * N_ + 2 * j + 1) * C_ + c];
    float cnt = 1.f;
    for (int s = 0; s < R_; ++s) {
        int si = edge[(size_t)b * HALF + s];
        int di = node_idx[(size_t)b * HALF + si];
        if (di == j) {
            val += xres[((size_t)b * N_ + 2 * si) * C_ + c];
            cnt += 1.f;
        }
    }
    xm[((size_t)b * NM + UNM + j) * C_ + c] = val / cnt;
}

// ======================= launcher =======================
extern "C" void kernel_launch(void* const* d_in, const int* in_sizes, int n_in,
                              void* d_out, int out_size, void* d_ws, size_t ws_size,
                              hipStream_t stream) {
    (void)in_sizes; (void)n_in; (void)out_size; (void)ws_size;

    const float*         x      = (const float*)d_in[0];
    const unsigned char* mask   = (const unsigned char*)d_in[1];
    const float*         ln1_g  = (const float*)d_in[2];
    const float*         ln1_b  = (const float*)d_in[3];
    const float*         w_qkv  = (const float*)d_in[4];
    const float*         b_qkv  = (const float*)d_in[5];
    const float*         w_proj = (const float*)d_in[6];
    const float*         b_proj = (const float*)d_in[7];
    const float*         ln2_g  = (const float*)d_in[8];
    const float*         ln2_b  = (const float*)d_in[9];
    const float*         w_fc1  = (const float*)d_in[10];
    const float*         b_fc1  = (const float*)d_in[11];
    const float*         w_fc2  = (const float*)d_in[12];
    const float*         b_fc2  = (const float*)d_in[13];
    float*               out    = (float*)d_out;

    // bump allocator over d_ws (256B aligned)
    uintptr_t base = (uintptr_t)d_ws;
    auto carve = [&](size_t bytes) -> uintptr_t {
        uintptr_t p = base;
        base += (bytes + 255) & ~(size_t)255;
        return p;
    };
    unsigned short* xn_bf16  = (unsigned short*)carve((size_t)M1 * C_ * 2);
    unsigned short* wqkvT    = (unsigned short*)carve((size_t)(3 * C_) * C_ * 2);
    unsigned short* wprojT   = (unsigned short*)carve((size_t)C_ * C_ * 2);
    unsigned short* wfc1T    = (unsigned short*)carve((size_t)HID_ * C_ * 2);
    unsigned short* wfc2T    = (unsigned short*)carve((size_t)C_ * HID_ * 2);
    float*          qkv      = (float*)carve((size_t)M1 * 3 * C_ * 4);
    unsigned short* o_bf16   = (unsigned short*)carve((size_t)M1 * C_ * 2);
    float*          xres     = (float*)carve((size_t)M1 * C_ * 4);
    float*          mnorm    = (float*)carve((size_t)B_ * N_ * DH_ * 4);
    float*          node_max = (float*)carve((size_t)B_ * HALF * 4);
    int*            node_idx = (int*)carve((size_t)B_ * HALF * 4);
    int*            edge     = (int*)carve((size_t)B_ * HALF * 4);
    float*          xm       = (float*)carve((size_t)M2 * C_ * 4);
    unsigned short* xn2_bf16 = (unsigned short*)carve((size_t)M2 * C_ * 2);
    unsigned short* h_bf16   = (unsigned short*)carve((size_t)M2 * HID_ * 2);

    // 1) weights -> bf16, transposed to N x K
    k_transpose_bf16<<<512, 256, 0, stream>>>(w_qkv,  wqkvT,  C_,  3 * C_);
    k_transpose_bf16<<<256, 256, 0, stream>>>(w_proj, wprojT, C_,  C_);
    k_transpose_bf16<<<512, 256, 0, stream>>>(w_fc1,  wfc1T,  C_,  HID_);
    k_transpose_bf16<<<512, 256, 0, stream>>>(w_fc2,  wfc2T,  HID_, C_);

    // 2) LN1 -> bf16 activations
    k_layernorm_bf16<<<M1, C_, 0, stream>>>(x, ln1_g, ln1_b, xn_bf16);

    // 3) QKV GEMM (WMMA): 8192x256 @ 256x768 -> qkv f32
    {
        dim3 grid((3 * C_) / TN, M1 / TM);
        k_gemm_nt_wmma<0, 0, 0><<<grid, 256, 0, stream>>>(
            xn_bf16, wqkvT, b_qkv, nullptr, qkv, nullptr, M1, 3 * C_, C_);
    }

    // 4) top-k attention -> o bf16
    k_topk_attn<<<B_ * H_ * N_, 128, 0, stream>>>(qkv, mask, o_bf16);

    // 5) metric (normalized k-mean)
    k_metric<<<B_ * N_, DH_, 0, stream>>>(qkv, mnorm);

    // 6) proj GEMM (WMMA) + residual x -> xres f32
    {
        dim3 grid(C_ / TN, M1 / TM);
        k_gemm_nt_wmma<0, 0, 1><<<grid, 256, 0, stream>>>(
            o_bf16, wprojT, b_proj, x, xres, nullptr, M1, C_, C_);
    }

    // 7) token merging
    k_simmax<<<B_ * HALF, 128, 0, stream>>>(mnorm, node_max, node_idx);
    k_rank<<<B_, HALF, 0, stream>>>(node_max, edge);
    k_merge_unm<<<B_ * UNM, C_, 0, stream>>>(xres, edge, xm);
    k_merge_dst<<<B_ * HALF, C_, 0, stream>>>(xres, edge, node_idx, xm);

    // 8) LN2 -> bf16
    k_layernorm_bf16<<<M2, C_, 0, stream>>>(xm, ln2_g, ln2_b, xn2_bf16);

    // 9) fc1 GEMM (WMMA) + GELU -> h bf16
    {
        dim3 grid(HID_ / TN, M2 / TM);
        k_gemm_nt_wmma<1, 1, 0><<<grid, 256, 0, stream>>>(
            xn2_bf16, wfc1T, b_fc1, nullptr, nullptr, h_bf16, M2, HID_, C_);
    }

    // 10) fc2 GEMM (WMMA) + residual xm -> out f32
    {
        dim3 grid(C_ / TN, M2 / TM);
        k_gemm_nt_wmma<0, 0, 1><<<grid, 256, 0, stream>>>(
            h_bf16, wfc2T, b_fc2, xm, out, nullptr, M2, C_, HID_);
    }
}
